// EdgeConvBlock_28295244546251
// MI455X (gfx1250) — compile-verified, matrix operand order checked by reading, not run on hardware
//
#include <hip/hip_runtime.h>

// ---------------------------------------------------------------------------
// EdgeConv block for MI455X (gfx1250, wave32, WMMA).
// Pipeline (all fused per 128-edge tile, bf16 WMMA w/ f32 accumulate):
//   pass0: z1 = [xi, xj-xi] @ W1            -> BN1 stats (LDS + global atomics)
//   pass1: h1 = lrelu(bn1(z1)); z2 = h1@W2  -> BN2 stats
//   skip : s  = x @ Ws                      -> BN_s stats, s cached in ws
//   pass2: h2 = lrelu(bn2(z2)) -> atomicMax (monotonic-uint f32) into agg[dst]
//   out  : lrelu(decode(agg) + bn_s(s)),  empty segments -> 0
// Wave layout: 8 waves split the 256 output channels (2 n-tiles each) and
// every wave covers all 8 m-subtiles -> each B-fragment is loaded exactly
// once per block and feeds 8 WMMAs (minimum L2 weight traffic).
// ---------------------------------------------------------------------------

typedef __attribute__((ext_vector_type(16))) __bf16 bf16x16;
typedef __attribute__((ext_vector_type(8)))  float  f32x8;

#define C_IN   128
#define O_CH   256
#define KT1    8          // GEMM1 k-tiles (2C=256 / 32)
#define KT2    8          // GEMM2 k-tiles (256 / 32)
#define KTS    4          // skip  k-tiles (128 / 32)
#define NT     16         // n-tiles (256 / 16)
#define MBLK   128        // rows (edges/nodes) per block
#define MSUB   8          // 16-row subtiles per block
#define SLOPE  0.2f
#define EPSF   1e-5f
#define ENC_NEGINF 0x007FFFFFu   // enc(-inf)

__device__ __forceinline__ unsigned enc_f32(float f) {
    unsigned u = __float_as_uint(f);
    return (u & 0x80000000u) ? ~u : (u | 0x80000000u);
}
__device__ __forceinline__ float dec_f32(unsigned e) {
    unsigned u = (e & 0x80000000u) ? (e & 0x7FFFFFFFu) : ~e;
    return __uint_as_float(u);
}
__device__ __forceinline__ f32x8 zero8() {
    f32x8 v = {0.f,0.f,0.f,0.f,0.f,0.f,0.f,0.f};
    return v;
}
__device__ __forceinline__ float lrelu(float z) { return z > 0.f ? z : SLOPE * z; }

// ---------------- weight packing into WMMA B-fragment order ----------------
// Bpack[kt][nt][lane][e] : per (kt,nt) a wave loads lane*32B contiguous.
// element (lane,e) holds W[k][n], k = kt*32 + (e<8?e:e+8) + (lane>=16?8:0),
// n = nt*16 + (lane&15).   (16-bit 32x16 B-matrix layout, ISA 7.12.2)
__global__ void k_pack(const float* __restrict__ W1, const float* __restrict__ W2,
                       const float* __restrict__ Ws,
                       __bf16* __restrict__ W1p, __bf16* __restrict__ W2p,
                       __bf16* __restrict__ Wsp)
{
    long long i = (long long)blockIdx.x * blockDim.x + threadIdx.x;
    const float* src; __bf16* dst; long long idx;
    if (i < 65536)       { src = W1; dst = W1p; idx = i; }
    else if (i < 131072) { src = W2; dst = W2p; idx = i - 65536; }
    else if (i < 163840) { src = Ws; dst = Wsp; idx = i - 131072; }
    else return;
    int e    = (int)(idx & 15);
    int lane = (int)((idx >> 4) & 31);
    int nt   = (int)((idx >> 9) & 15);
    int kt   = (int)(idx >> 13);
    int k = kt * 32 + (e < 8 ? e : e + 8) + ((lane >= 16) ? 8 : 0);
    int n = nt * 16 + (lane & 15);
    dst[idx] = (__bf16)src[(long long)k * O_CH + n];
}

// ---------------- init: agg = enc(-inf), BN accumulators = 0 ---------------
__global__ void k_init(unsigned* __restrict__ agg, float* __restrict__ stats,
                       long long aggN)
{
    long long i = (long long)blockIdx.x * blockDim.x + threadIdx.x;
    if (i < aggN) agg[i] = ENC_NEGINF;
    if (i < 6 * 256) stats[i] = 0.f;
}

// ---------------- BN finalize: scale = g*rstd, shift = b - mean*scale ------
__global__ void k_finalize(const float* __restrict__ sum, const float* __restrict__ sq,
                           const float* __restrict__ g, const float* __restrict__ b,
                           float* __restrict__ scale, float* __restrict__ shift,
                           float cnt)
{
    int ch = threadIdx.x;
    float mean = sum[ch] / cnt;
    float var  = fmaxf(sq[ch] / cnt - mean * mean, 0.f);
    float sc   = g[ch] * rsqrtf(var + EPSF);
    scale[ch] = sc;
    shift[ch] = b[ch] - mean * sc;
}

// ---------------- fused per-edge pass --------------------------------------
// mode 0: z1 only, accumulate BN1 stats
// mode 1: z1 -> bn1/lrelu -> h1 -> z2, accumulate BN2 stats
// mode 2: full chain, bn2/lrelu, atomicMax into agg[dst]
__global__ __launch_bounds__(256)
void k_edge_pass(const float* __restrict__ x, const int* __restrict__ ei,
                 const __bf16* __restrict__ W1p, const __bf16* __restrict__ W2p,
                 float* __restrict__ sum1, float* __restrict__ sq1,
                 float* __restrict__ sum2, float* __restrict__ sq2,
                 const float* __restrict__ sc1, const float* __restrict__ sh1,
                 const float* __restrict__ sc2, const float* __restrict__ sh2,
                 unsigned* __restrict__ agg, int E, int mode)
{
    __shared__ __bf16 mfrag[MSUB * KT1 * 32 * 16];   // 64 KB, A-frag order
    __shared__ __bf16 hfrag[MSUB * KT2 * 32 * 16];   // 64 KB, A-frag order

    const int tid  = threadIdx.x;
    const int lane = tid & 31;
    const int ng   = tid >> 5;             // 0..7 : n-tile group (2 ntiles)
    const long long tile0 = (long long)blockIdx.x * MBLK;
    const int hi8 = (lane >= 16) ? 8 : 0;

    // ---- stage m = [xi, xj-xi] (bf16) directly in A-fragment order ----
    for (int idx = tid; idx < MSUB * KT1 * 32 * 2; idx += 256) {
        int half = idx & 1;
        int ln   = (idx >> 1) & 31;
        int kt   = (idx >> 6) & 7;
        int ms   = (idx >> 9) & 7;
        long long e = tile0 + ms * 16 + (ln & 15);
        if (e >= E) e = E - 1;
        long long ns = (long long)ei[e];                  // src
        long long nd = (long long)ei[(long long)E + e];   // dst
        int kbase = kt * 32 + ((ln >= 16) ? 8 : 0) + half * 16;
        __bf16* dp = &mfrag[((ms * KT1 + kt) * 32 + ln) * 16 + half * 8];
        if (kbase < C_IN) {
            const float* xi = x + nd * C_IN + kbase;
            #pragma unroll
            for (int t = 0; t < 8; ++t) dp[t] = (__bf16)xi[t];
        } else {
            const float* xi = x + nd * C_IN + (kbase - C_IN);
            const float* xj = x + ns * C_IN + (kbase - C_IN);
            #pragma unroll
            for (int t = 0; t < 8; ++t) dp[t] = (__bf16)(xj[t] - xi[t]);
        }
    }
    __syncthreads();

    // ---- GEMM1: z1 = m @ W1 (per wave: 8 msub x 2 ntile = 16 WMMA/ktile) --
    f32x8 acc1[MSUB][2];
    #pragma unroll
    for (int i = 0; i < MSUB; ++i)
        #pragma unroll
        for (int j = 0; j < 2; ++j) acc1[i][j] = zero8();

    for (int kt = 0; kt < KT1; ++kt) {
        bf16x16 a[MSUB];
        #pragma unroll
        for (int i = 0; i < MSUB; ++i)
            a[i] = *(const bf16x16*)&mfrag[((i * KT1 + kt) * 32 + lane) * 16];
        #pragma unroll
        for (int j = 0; j < 2; ++j) {
            int nt = ng * 2 + j;
            bf16x16 bfr = *(const bf16x16*)&W1p[((kt * NT + nt) * 32 + lane) * 16];
            #pragma unroll
            for (int i = 0; i < MSUB; ++i)
                acc1[i][j] = __builtin_amdgcn_wmma_f32_16x16x32_bf16(false, a[i],
                                 false, bfr, (short)0, acc1[i][j], false, false);
        }
    }

    if (mode == 0) {
        // ---- BN1 stats: per-channel sum / sumsq via LDS then global ----
        __syncthreads();                       // mfrag reads done
        float* red = (float*)mfrag;            // 512 floats
        for (int i = tid; i < 512; i += 256) red[i] = 0.f;
        __syncthreads();
        #pragma unroll
        for (int j = 0; j < 2; ++j) {
            int ch = (ng * 2 + j) * 16 + (lane & 15);
            float s = 0.f, s2 = 0.f;
            #pragma unroll
            for (int i = 0; i < MSUB; ++i)
                #pragma unroll
                for (int v = 0; v < 8; ++v) {
                    long long e = tile0 + i * 16 + v + hi8;
                    if (e < E) { float z = acc1[i][j][v]; s += z; s2 += z * z; }
                }
            atomicAdd(&red[ch], s);
            atomicAdd(&red[256 + ch], s2);
        }
        __syncthreads();
        for (int i = tid; i < 256; i += 256) {
            atomicAdd(&sum1[i], red[i]);
            atomicAdd(&sq1[i], red[256 + i]);
        }
        return;
    }

    // ---- bn1 + lrelu -> h1, written in A-fragment order for GEMM2 ----
    #pragma unroll
    for (int j = 0; j < 2; ++j) {
        int ch = (ng * 2 + j) * 16 + (lane & 15);
        float scl = sc1[ch], shf = sh1[ch];
        int kt = ch >> 5, r = ch & 31;
        int chi, e8;
        if (r < 8)       { chi = 0; e8 = r; }
        else if (r < 16) { chi = 1; e8 = r - 8; }
        else if (r < 24) { chi = 0; e8 = r - 8; }
        else             { chi = 1; e8 = r - 16; }
        #pragma unroll
        for (int i = 0; i < MSUB; ++i)
            #pragma unroll
            for (int v = 0; v < 8; ++v) {
                int Ml = v + hi8;                         // local row 0..15
                float z = lrelu(acc1[i][j][v] * scl + shf);
                hfrag[((i * KT2 + kt) * 32 + (Ml + chi * 16)) * 16 + e8] = (__bf16)z;
            }
    }
    __syncthreads();

    // ---- GEMM2: z2 = h1 @ W2 ----
    f32x8 acc2[MSUB][2];
    #pragma unroll
    for (int i = 0; i < MSUB; ++i)
        #pragma unroll
        for (int j = 0; j < 2; ++j) acc2[i][j] = zero8();

    for (int kt = 0; kt < KT2; ++kt) {
        bf16x16 a[MSUB];
        #pragma unroll
        for (int i = 0; i < MSUB; ++i)
            a[i] = *(const bf16x16*)&hfrag[((i * KT2 + kt) * 32 + lane) * 16];
        #pragma unroll
        for (int j = 0; j < 2; ++j) {
            int nt = ng * 2 + j;
            bf16x16 bfr = *(const bf16x16*)&W2p[((kt * NT + nt) * 32 + lane) * 16];
            #pragma unroll
            for (int i = 0; i < MSUB; ++i)
                acc2[i][j] = __builtin_amdgcn_wmma_f32_16x16x32_bf16(false, a[i],
                                 false, bfr, (short)0, acc2[i][j], false, false);
        }
    }

    if (mode == 1) {
        // ---- BN2 stats over raw z2 ----
        __syncthreads();
        float* red = (float*)mfrag;            // mfrag free after GEMM1
        for (int i = tid; i < 512; i += 256) red[i] = 0.f;
        __syncthreads();
        #pragma unroll
        for (int j = 0; j < 2; ++j) {
            int ch = (ng * 2 + j) * 16 + (lane & 15);
            float s = 0.f, s2 = 0.f;
            #pragma unroll
            for (int i = 0; i < MSUB; ++i)
                #pragma unroll
                for (int v = 0; v < 8; ++v) {
                    long long e = tile0 + i * 16 + v + hi8;
                    if (e < E) { float z = acc2[i][j][v]; s += z; s2 += z * z; }
                }
            atomicAdd(&red[ch], s);
            atomicAdd(&red[256 + ch], s2);
        }
        __syncthreads();
        for (int i = tid; i < 256; i += 256) {
            atomicAdd(&sum2[i], red[i]);
            atomicAdd(&sq2[i], red[256 + i]);
        }
        return;
    }

    // ---- mode 2: bn2 + lrelu -> segment max (monotonic-uint atomicMax) ----
    float scj[2], shj[2];
    #pragma unroll
    for (int j = 0; j < 2; ++j) {
        int ch = (ng * 2 + j) * 16 + (lane & 15);
        scj[j] = sc2[ch]; shj[j] = sh2[ch];
    }
    #pragma unroll
    for (int i = 0; i < MSUB; ++i)
        #pragma unroll
        for (int v = 0; v < 8; ++v) {
            long long e = tile0 + i * 16 + v + hi8;
            if (e >= E) continue;
            long long nd = (long long)ei[(long long)E + e];
            unsigned* arow = agg + nd * O_CH;
            #pragma unroll
            for (int j = 0; j < 2; ++j) {
                int ch = (ng * 2 + j) * 16 + (lane & 15);
                float z = lrelu(acc2[i][j][v] * scj[j] + shj[j]);
                atomicMax(&arow[ch], enc_f32(z));
            }
        }
}

// ---------------- skip path: s = x @ Ws (raw), BN_s stats ------------------
__global__ __launch_bounds__(256)
void k_skip_pass(const float* __restrict__ x, const __bf16* __restrict__ Wsp,
                 float* __restrict__ sums, float* __restrict__ sqs,
                 float* __restrict__ sbuf, int N)
{
    __shared__ __bf16 xfrag[MSUB * KTS * 32 * 16];   // 32 KB

    const int tid  = threadIdx.x;
    const int lane = tid & 31;
    const int ng   = tid >> 5;             // 0..7
    const long long tile0 = (long long)blockIdx.x * MBLK;
    const int hi8 = (lane >= 16) ? 8 : 0;

    for (int idx = tid; idx < MSUB * KTS * 32 * 2; idx += 256) {
        int half = idx & 1;
        int ln   = (idx >> 1) & 31;
        int kt   = (idx >> 6) & 3;
        int ms   = (idx >> 8) & 7;
        long long n = tile0 + ms * 16 + (ln & 15);
        if (n >= N) n = N - 1;
        int kbase = kt * 32 + ((ln >= 16) ? 8 : 0) + half * 16;
        const float* xr = x + n * C_IN + kbase;
        __bf16* dp = &xfrag[((ms * KTS + kt) * 32 + ln) * 16 + half * 8];
        #pragma unroll
        for (int t = 0; t < 8; ++t) dp[t] = (__bf16)xr[t];
    }
    __syncthreads();

    f32x8 acc[MSUB][2];
    #pragma unroll
    for (int i = 0; i < MSUB; ++i)
        #pragma unroll
        for (int j = 0; j < 2; ++j) acc[i][j] = zero8();

    for (int kt = 0; kt < KTS; ++kt) {
        bf16x16 a[MSUB];
        #pragma unroll
        for (int i = 0; i < MSUB; ++i)
            a[i] = *(const bf16x16*)&xfrag[((i * KTS + kt) * 32 + lane) * 16];
        #pragma unroll
        for (int j = 0; j < 2; ++j) {
            int nt = ng * 2 + j;
            bf16x16 bfr = *(const bf16x16*)&Wsp[((kt * NT + nt) * 32 + lane) * 16];
            #pragma unroll
            for (int i = 0; i < MSUB; ++i)
                acc[i][j] = __builtin_amdgcn_wmma_f32_16x16x32_bf16(false, a[i],
                                false, bfr, (short)0, acc[i][j], false, false);
        }
    }

    __syncthreads();
    float* red = (float*)xfrag;
    for (int i = tid; i < 512; i += 256) red[i] = 0.f;
    __syncthreads();
    #pragma unroll
    for (int j = 0; j < 2; ++j) {
        int ch = (ng * 2 + j) * 16 + (lane & 15);
        float s = 0.f, s2 = 0.f;
        #pragma unroll
        for (int i = 0; i < MSUB; ++i)
            #pragma unroll
            for (int v = 0; v < 8; ++v) {
                long long n = tile0 + i * 16 + v + hi8;
                if (n < N) {
                    float z = acc[i][j][v];
                    sbuf[n * O_CH + ch] = z;          // raw; BN applied in k_output
                    s += z; s2 += z * z;
                }
            }
        atomicAdd(&red[ch], s);
        atomicAdd(&red[256 + ch], s2);
    }
    __syncthreads();
    for (int i = tid; i < 256; i += 256) {
        atomicAdd(&sums[i], red[i]);
        atomicAdd(&sqs[i], red[256 + i]);
    }
}

// ---------------- output: lrelu(decode(agg) + bn_s(s)) ---------------------
__global__ void k_output(const unsigned* __restrict__ agg, const float* __restrict__ sbuf,
                         const float* __restrict__ scs, const float* __restrict__ shs,
                         float* __restrict__ out, long long total)
{
    long long i = (long long)blockIdx.x * blockDim.x + threadIdx.x;
    if (i >= total) return;
    int ch = (int)(i & 255);
    unsigned e = agg[i];
    float a = (e == ENC_NEGINF) ? 0.f : dec_f32(e);   // empty segment -> 0
    float z = a + sbuf[i] * scs[ch] + shs[ch];
    out[i] = lrelu(z);
}

// ---------------------------------------------------------------------------
extern "C" void kernel_launch(void* const* d_in, const int* in_sizes, int n_in,
                              void* d_out, int out_size, void* d_ws, size_t ws_size,
                              hipStream_t stream) {
    const float* x  = (const float*)d_in[0];
    const int*   ei = (const int*)d_in[1];
    const float* W1 = (const float*)d_in[2];
    const float* g1 = (const float*)d_in[3];
    const float* b1 = (const float*)d_in[4];
    const float* W2 = (const float*)d_in[5];
    const float* g2 = (const float*)d_in[6];
    const float* b2 = (const float*)d_in[7];
    const float* Ws = (const float*)d_in[8];
    const float* gs = (const float*)d_in[9];
    const float* bs = (const float*)d_in[10];

    const int N = in_sizes[0] / C_IN;     // 50000
    const int E = in_sizes[1] / 2;        // 800000
    float* out = (float*)d_out;

    char* ws = (char*)d_ws;
    __bf16* W1p = (__bf16*)(ws);                       // 128 KB
    __bf16* W2p = (__bf16*)(ws + 131072);              // 128 KB
    __bf16* Wsp = (__bf16*)(ws + 262144);              //  64 KB
    float*  st  = (float*) (ws + 327680);              //  12 KB of stats
    float *sum1 = st,        *sq1 = st + 256;
    float *sum2 = st + 512,  *sq2 = st + 768;
    float *sums = st + 1024, *sqs = st + 1280;
    float *sc1  = st + 1536, *sh1 = st + 1792;
    float *sc2  = st + 2048, *sh2 = st + 2304;
    float *scs  = st + 2560, *shs = st + 2816;
    unsigned* agg = (unsigned*)(ws + 327680 + 12288);
    float* sbuf = (float*)(ws + 327680 + 12288 + (size_t)N * O_CH * 4);

    const long long aggN = (long long)N * O_CH;
    const int eblocks = (E + MBLK - 1) / MBLK;
    const int nblocks = (N + MBLK - 1) / MBLK;

    k_init<<<(unsigned)((aggN + 255) / 256), 256, 0, stream>>>(agg, st, aggN);
    k_pack<<<640, 256, 0, stream>>>(W1, W2, Ws, W1p, W2p, Wsp);

    // pass 0: BN1 stats
    k_edge_pass<<<eblocks, 256, 0, stream>>>(x, ei, W1p, W2p, sum1, sq1, sum2, sq2,
                                             sc1, sh1, sc2, sh2, agg, E, 0);
    k_finalize<<<1, 256, 0, stream>>>(sum1, sq1, g1, b1, sc1, sh1, (float)E);
    // pass 1: BN2 stats
    k_edge_pass<<<eblocks, 256, 0, stream>>>(x, ei, W1p, W2p, sum1, sq1, sum2, sq2,
                                             sc1, sh1, sc2, sh2, agg, E, 1);
    k_finalize<<<1, 256, 0, stream>>>(sum2, sq2, g2, b2, sc2, sh2, (float)E);
    // skip path + its BN stats
    k_skip_pass<<<nblocks, 256, 0, stream>>>(x, Wsp, sums, sqs, sbuf, N);
    k_finalize<<<1, 256, 0, stream>>>(sums, sqs, gs, bs, scs, shs, (float)N);
    // pass 2: full chain + segment max
    k_edge_pass<<<eblocks, 256, 0, stream>>>(x, ei, W1p, W2p, sum1, sq1, sum2, sq2,
                                             sc1, sh1, sc2, sh2, agg, E, 2);
    // combine
    k_output<<<(unsigned)((aggN + 255) / 256), 256, 0, stream>>>(agg, sbuf, scs, shs,
                                                                 out, aggN);
}
